// contextual_loss_39960375722309
// MI455X (gfx1250) — matrix operand order repaired and checked by reading, not compiled, not run on hardware
//
#include <hip/hip_runtime.h>
#include <hip/hip_bf16.h>

#define NB   4
#define CH   256
#define HW   4096
#define TILE 128
#define LDSROW 264          // 256 bf16 + 8 pad per row
#define EPSF 1e-5f

typedef __attribute__((ext_vector_type(16))) __bf16 v16bf;
typedef __attribute__((ext_vector_type(8)))  float  v8f;

// monotonic float <-> uint mapping so atomicMin/atomicMax on uints give
// order-independent (deterministic) float min/max
__device__ __forceinline__ unsigned fenc(float f){
  unsigned u = __float_as_uint(f);
  return (u & 0x80000000u) ? ~u : (u | 0x80000000u);
}
__device__ __forceinline__ float fdec(unsigned u){
  return __uint_as_float((u & 0x80000000u) ? (u & 0x7FFFFFFFu) : ~u);
}
__device__ __forceinline__ unsigned short f2bf(float f){
  unsigned u = __float_as_uint(f);
  u += 0x7FFFu + ((u >> 16) & 1u);          // round-to-nearest-even
  return (unsigned short)(u >> 16);
}

// ---------------- per-channel mean of y ----------------
__global__ void mu_kernel(const float* __restrict__ y, float* __restrict__ mu){
  __shared__ float red[256];
  const int c = blockIdx.x, t = threadIdx.x;
  float s = 0.f;
  for (int i = t; i < NB*HW; i += 256){
    int nn = i >> 12, p = i & (HW-1);
    s += y[((size_t)nn*CH + c)*HW + p];
  }
  red[t] = s; __syncthreads();
  for (int o = 128; o > 0; o >>= 1){ if (t < o) red[t] += red[t+o]; __syncthreads(); }
  if (t == 0) mu[c] = red[0] * (1.0f/(NB*HW));
}

// ---------------- init scratch ----------------
__global__ void init_kernel(unsigned* __restrict__ dmin, unsigned* __restrict__ colmax,
                            float* __restrict__ partial){
  int i = blockIdx.x*256 + threadIdx.x;
  if (i < NB*HW){ dmin[i] = 0xFFFFFFFFu; colmax[i] = 0u; }
  if (i < NB*32*HW) partial[i] = 0.f;
}

// ---------------- center + L2 normalize -> bf16 (n,p,c) layout ----------------
__global__ void norm_kernel(const float* __restrict__ x, const float* __restrict__ y,
                            const float* __restrict__ mu,
                            unsigned short* __restrict__ xn, unsigned short* __restrict__ yn){
  __shared__ float mu_s[CH];
  __shared__ float sxs[8][33], sys[8][33];
  __shared__ float rix[32], riy[32];
  const int t = threadIdx.x;
  const int n = blockIdx.y, pb = blockIdx.x * 32;
  mu_s[t] = mu[t];
  __syncthreads();
  // phase 1: coalesced reads (lanes over positions), sum-of-squares per position
  const int pos = t & 31, cg = t >> 5;
  float sx = 0.f, sy = 0.f;
  for (int c = cg; c < CH; c += 8){
    float m = mu_s[c];
    size_t idx = ((size_t)(n*CH + c))*HW + pb + pos;
    float xv = x[idx] - m, yv = y[idx] - m;
    sx += xv*xv; sy += yv*yv;
  }
  sxs[cg][pos] = sx; sys[cg][pos] = sy;
  __syncthreads();
  if (t < 32){
    float ax = 0.f, ay = 0.f;
    for (int g = 0; g < 8; ++g){ ax += sxs[g][t]; ay += sys[g][t]; }
    rix[t] = rsqrtf(ax); riy[t] = rsqrtf(ay);
  }
  __syncthreads();
  // phase 2: contiguous bf16 writes (thread owns a 64B span of one row)
  const int pos2 = t >> 3, cs = t & 7;
  const float rx = rix[pos2], ry = riy[pos2];
  const size_t ob = ((size_t)(n*HW + pb + pos2))*CH;
  const size_t ib = ((size_t)n*CH)*HW + pb + pos2;
  for (int c = cs*32; c < cs*32 + 32; c += 2){
    float x0 = (x[ib + (size_t)c*HW]     - mu_s[c])   * rx;
    float x1 = (x[ib + (size_t)(c+1)*HW] - mu_s[c+1]) * rx;
    float y0 = (y[ib + (size_t)c*HW]     - mu_s[c])   * ry;
    float y1 = (y[ib + (size_t)(c+1)*HW] - mu_s[c+1]) * ry;
    *(unsigned*)(xn + ob + c) = (unsigned)f2bf(x0) | ((unsigned)f2bf(x1) << 16);
    *(unsigned*)(yn + ob + c) = (unsigned)f2bf(y0) | ((unsigned)f2bf(y1) << 16);
  }
}

// ---------------- fused WMMA GEMM + reduction passes ----------------
// MODE 0: row-min of d=1-cos   -> dmin (atomicMin on encoded floats)
// MODE 1: row-sum of w         -> per-colblock partials (2 commutative atomicAdds)
// MODE 2: col-max of w/rowsum  -> colmax (atomicMax on encoded floats)
template<int MODE>
__global__ __launch_bounds__(256) void gemm_kernel(
    const unsigned short* __restrict__ xn, const unsigned short* __restrict__ yn,
    unsigned* __restrict__ dmin, float* __restrict__ partial,
    const float* __restrict__ rowsum, unsigned* __restrict__ colmax)
{
  __shared__ __align__(16) unsigned short lA[TILE*LDSROW];
  __shared__ __align__(16) unsigned short lB[TILE*LDSROW];
  __shared__ float rsA[TILE];
  __shared__ float rsB[TILE];

  const int t = threadIdx.x;
  const int n = blockIdx.z;
  const int pBase = blockIdx.y * TILE;
  const int qBase = blockIdx.x * TILE;

  // stage full 128x256 bf16 strips for A and B into LDS (b128 copies)
  {
    const int row = t >> 1, half = t & 1;
    const uint4* gA = (const uint4*)(xn + ((size_t)(n*HW + pBase + row))*CH + half*128);
    const uint4* gB = (const uint4*)(yn + ((size_t)(n*HW + qBase + row))*CH + half*128);
    uint4* sA = (uint4*)(lA + row*LDSROW + half*128);
    uint4* sB = (uint4*)(lB + row*LDSROW + half*128);
    #pragma unroll
    for (int i = 0; i < 16; ++i){ sA[i] = gA[i]; sB[i] = gB[i]; }
  }
  __syncthreads();

  const int lane = t & 31, wv = t >> 5;
  const int wr = wv & 3, wc = wv >> 2;      // 4 row-waves x 2 col-waves
  const int rB = wr*32, cB = wc*64;
  const int lm = lane & 15, lh = lane >> 4;

  v8f acc[2][4];
  #pragma unroll
  for (int a = 0; a < 2; ++a)
    #pragma unroll
    for (int b = 0; b < 4; ++b)
      acc[a][b] = (v8f){0.f,0.f,0.f,0.f,0.f,0.f,0.f,0.f};

  union Frag { uint4 u[2]; v16bf v; };

  #pragma unroll
  for (int kk = 0; kk < 8; ++kk){
    const int kb = kk*32;
    Frag fa[2], fb[4];
    #pragma unroll
    for (int tr = 0; tr < 2; ++tr){
      // A 16x32 bf16 layout: lanes 0-15 K[kb..kb+7],[kb+16..kb+23]; lanes 16-31 shifted by 8
      const unsigned short* base = lA + (rB + tr*16 + lm)*LDSROW + kb + lh*8;
      fa[tr].u[0] = *(const uint4*)base;
      fa[tr].u[1] = *(const uint4*)(base + 16);
    }
    #pragma unroll
    for (int tc = 0; tc < 4; ++tc){
      // B 32x16 bf16 layout: lanes 0-15 K[kb..kb+7],[kb+16..kb+23]; lanes 16-31 shifted by 8
      const unsigned short* base = lB + (cB + tc*16 + lm)*LDSROW + kb + lh*8;
      fb[tc].u[0] = *(const uint4*)base;
      fb[tc].u[1] = *(const uint4*)(base + 16);
    }
    #pragma unroll
    for (int tr = 0; tr < 2; ++tr)
      #pragma unroll
      for (int tc = 0; tc < 4; ++tc)
        acc[tr][tc] = __builtin_amdgcn_wmma_f32_16x16x32_bf16(
            false, fa[tr].v, false, fb[tc].v, (short)0, acc[tr][tc], false, false);
  }

  // C/D layout: VGPR v -> M = v (lanes 0-15) or v+8 (lanes 16-31), N = lane%16
  if (MODE == 0){
    #pragma unroll
    for (int tr = 0; tr < 2; ++tr){
      #pragma unroll
      for (int v = 0; v < 8; ++v){
        float dv =        1.0f - acc[tr][0][v];
        dv = fminf(dv, 1.0f - acc[tr][1][v]);
        dv = fminf(dv, 1.0f - acc[tr][2][v]);
        dv = fminf(dv, 1.0f - acc[tr][3][v]);
        #pragma unroll
        for (int off = 1; off < 16; off <<= 1)
          dv = fminf(dv, __shfl_xor(dv, off, 32));
        if (lm == 0){
          int r = pBase + rB + tr*16 + v + lh*8;
          atomicMin(&dmin[n*HW + r], fenc(dv));
        }
      }
    }
  }

  if (MODE == 1){
    if (t < TILE) rsA[t] = 2.0f / (fdec(dmin[n*HW + pBase + t]) + EPSF);
    __syncthreads();
    #pragma unroll
    for (int tr = 0; tr < 2; ++tr){
      #pragma unroll
      for (int v = 0; v < 8; ++v){
        const int r = rB + tr*16 + v + lh*8;
        const float ar = rsA[r];
        float s = 0.f;
        #pragma unroll
        for (int tc = 0; tc < 4; ++tc)
          s += __expf(2.0f - ar * (1.0f - acc[tr][tc][v]));
        #pragma unroll
        for (int off = 1; off < 16; off <<= 1)
          s += __shfl_xor(s, off, 32);
        if (lm == 0)
          atomicAdd(&partial[((size_t)(n*32 + blockIdx.x))*HW + pBase + r], s);
      }
    }
  }

  if (MODE == 2){
    if (t < TILE){
      rsA[t] = 2.0f / (fdec(dmin[n*HW + pBase + t]) + EPSF);
      rsB[t] = 1.0f / rowsum[n*HW + pBase + t];
    }
    __syncthreads();
    #pragma unroll
    for (int tc = 0; tc < 4; ++tc){
      float m = 0.f;
      #pragma unroll
      for (int tr = 0; tr < 2; ++tr)
        #pragma unroll
        for (int v = 0; v < 8; ++v){
          const int r = rB + tr*16 + v + lh*8;
          float cx = __expf(2.0f - rsA[r]*(1.0f - acc[tr][tc][v])) * rsB[r];
          m = fmaxf(m, cx);
        }
      m = fmaxf(m, __shfl_xor(m, 16, 32));
      if (lh == 0)
        atomicMax(&colmax[n*HW + qBase + cB + tc*16 + lm], fenc(m));
    }
  }
}

// ---------------- reduce colblock partial row-sums (fixed order) ----------------
__global__ void rowsum_kernel(const float* __restrict__ partial, float* __restrict__ rowsum){
  int i = blockIdx.x*256 + threadIdx.x;     // 0..16383 -> (n,p)
  int n = i >> 12, p = i & (HW-1);
  float s = 0.f;
  for (int cb = 0; cb < 32; ++cb)
    s += partial[((size_t)(n*32 + cb))*HW + p];
  rowsum[i] = s;
}

// ---------------- final loss ----------------
__global__ void final_kernel(const unsigned* __restrict__ colmax, float* __restrict__ out){
  __shared__ float red[256];
  const int t = threadIdx.x;
  float loss = 0.f;
  for (int n = 0; n < NB; ++n){
    float s = 0.f;
    for (int q = t; q < HW; q += 256) s += fdec(colmax[n*HW + q]);
    red[t] = s; __syncthreads();
    for (int o = 128; o > 0; o >>= 1){ if (t < o) red[t] += red[t+o]; __syncthreads(); }
    if (t == 0) loss += -logf(red[0]*(1.0f/HW) + 1e-5f);
    __syncthreads();
  }
  if (t == 0) out[0] = loss * (1.0f/NB);
}

extern "C" void kernel_launch(void* const* d_in, const int* in_sizes, int n_in,
                              void* d_out, int out_size, void* d_ws, size_t ws_size,
                              hipStream_t stream){
  const float* x = (const float*)d_in[0];
  const float* y = (const float*)d_in[1];
  float* out = (float*)d_out;

  char* ws = (char*)d_ws;
  size_t off = 0;
  auto alloc = [&](size_t bytes)->char*{
    char* p = ws + off;
    off = (off + bytes + 255) & ~(size_t)255;
    return p;
  };
  float*          mu      = (float*)alloc((size_t)CH*4);
  unsigned*       dmin    = (unsigned*)alloc((size_t)NB*HW*4);
  float*          rowsum  = (float*)alloc((size_t)NB*HW*4);
  unsigned*       colmax  = (unsigned*)alloc((size_t)NB*HW*4);
  float*          partial = (float*)alloc((size_t)NB*32*HW*4);
  unsigned short* xn      = (unsigned short*)alloc((size_t)NB*HW*CH*2);
  unsigned short* yn      = (unsigned short*)alloc((size_t)NB*HW*CH*2);

  mu_kernel<<<CH, 256, 0, stream>>>(y, mu);
  init_kernel<<<(NB*32*HW)/256, 256, 0, stream>>>(dmin, colmax, partial);
  norm_kernel<<<dim3(HW/32, NB), 256, 0, stream>>>(x, y, mu, xn, yn);

  dim3 g(HW/TILE, HW/TILE, NB);
  gemm_kernel<0><<<g, 256, 0, stream>>>(xn, yn, dmin, partial, rowsum, colmax);
  gemm_kernel<1><<<g, 256, 0, stream>>>(xn, yn, dmin, partial, rowsum, colmax);
  rowsum_kernel<<<(NB*HW)/256, 256, 0, stream>>>(partial, rowsum);
  gemm_kernel<2><<<g, 256, 0, stream>>>(xn, yn, dmin, partial, rowsum, colmax);
  final_kernel<<<1, 256, 0, stream>>>(colmax, out);
}
